// Deformable_Conv_Block_12610023981069
// MI455X (gfx1250) — matrix operand order repaired and checked by reading, not compiled, not run on hardware
//
#include <hip/hip_runtime.h>

typedef __attribute__((ext_vector_type(16))) _Float16     v16h;
typedef __attribute__((ext_vector_type(8)))  float        v8f;
typedef __attribute__((ext_vector_type(4)))  unsigned int v4u;
typedef __attribute__((ext_vector_type(8)))  int          v8i;
typedef __attribute__((ext_vector_type(4)))  int          v4i;

#define HH 160
#define WW 160
#define BB 8
#define CINC 64

// ---------------------------------------------------------------------------
// Pack OIHW f32 weights into exact WMMA A-fragment layout (16x32 f16, M=Cout,
// K=Cin).  fr = ((ky*3+kx)*2 + kb)*MB + mb ; per lane 16 contiguous halfs.
// ---------------------------------------------------------------------------
__global__ void pack_w_kernel(const float* __restrict__ w,
                              _Float16* __restrict__ dst, int cout) {
  int t = blockIdx.x * blockDim.x + threadIdx.x;
  int MBn = cout >> 4;
  int total = 9 * 2 * MBn * 32;
  if (t >= total) return;
  int lane = t & 31;
  int fr = t >> 5;
  int mb = fr % MBn;
  int kb = (fr / MBn) & 1;
  int tap = fr / (2 * MBn);                 // ky*3+kx
  int o = mb * 16 + (lane & 15);
  int hs = (lane >> 4) ? 8 : 0;
  _Float16* d = dst + ((size_t)fr * 32 + lane) * 16;
  for (int e = 0; e < 16; ++e) {
    int K = (lane >> 4) * 16 + e;
    int i = kb * 32 + K;
    d[e] = (_Float16)w[((size_t)o * CINC + i) * 9 + tap];
  }
}

// NCHW f32 -> NHWC f16 via LDS tile transpose (coalesced both sides).
__global__ __launch_bounds__(256) void nchw_to_nhwc_f16(
    const float* __restrict__ in, _Float16* __restrict__ out) {
  __shared__ _Float16 tile[CINC * 32];
  const int w0 = blockIdx.x * 32;
  const int h = blockIdx.y;
  const int b = blockIdx.z;
  const int tid = threadIdx.x;
  for (int i = tid; i < CINC * 32; i += 256) {   // i = c*32 + w'
    int c = i >> 5, wp = i & 31;
    tile[i] = (_Float16)in[(((size_t)b * CINC + c) * HH + h) * WW + w0 + wp];
  }
  __syncthreads();
  _Float16* dst = out + ((size_t)(b * HH + h) * WW + w0) * CINC;
  for (int i = tid; i < CINC * 32; i += 256) {   // i = w'*64 + c
    int wp = i >> 6, c = i & 63;
    dst[(size_t)wp * CINC + c] = tile[c * 32 + wp];
  }
}

// ---------------------------------------------------------------------------
// TDM: DMA one input row (160*64 f16 = 2560 x 8B elements, 1-D tile) into LDS.
// D# per CDNA5 ISA; 6-arg (clang-23) builtin.
// ---------------------------------------------------------------------------
__device__ __forceinline__ void tdm_load_row(const _Float16* gsrc,
                                             unsigned lds_off) {
  unsigned long long ga = (unsigned long long)(uintptr_t)gsrc;
  v4u g0;
  g0[0] = 1u;                                     // count=1, user descriptor
  g0[1] = lds_off;                                // lds_addr (bytes)
  g0[2] = (unsigned)(ga & 0xFFFFFFFFu);           // global_addr[31:0]
  g0[3] = (unsigned)((ga >> 32) & 0x01FFFFFFu)    // global_addr[56:32]
          | (2u << 30);                           // type = 2 ("image")
  v8i g1;
  g1[0] = (int)(3u << 16);                        // data_size = 3 (8 bytes)
  g1[1] = (int)((2560u & 0xFFFFu) << 16);         // tensor_dim0[15:0]
  g1[2] = (int)((2560u >> 16) | (1u << 16));      // tensor_dim0[31:16]|dim1=1
  g1[3] = (int)(2560u << 16);                     // tile_dim0 = 2560
  g1[4] = 0;                                      // tile_dim1/2 unused
  g1[5] = (int)2560u;                             // tensor_dim0_stride
  g1[6] = 0;
  g1[7] = 0;
  v4i z4 = {0, 0, 0, 0};
  v8i z8 = {0, 0, 0, 0, 0, 0, 0, 0};
  __builtin_amdgcn_tensor_load_to_lds(g0, g1, z4, z4, z8, 0);
}

__device__ __forceinline__ void wait_tensor_le(int n) {
  switch (n) {
    case 0: __builtin_amdgcn_s_wait_tensorcnt(0); break;
    case 1: __builtin_amdgcn_s_wait_tensorcnt(1); break;
    default: __builtin_amdgcn_s_wait_tensorcnt(2); break;
  }
}

// ---------------------------------------------------------------------------
// 3x3 SAME conv as implicit GEMM over WMMA.  One block = (row h, batch b).
// Waves are tiled (MB x 2): each wave owns ONE Cout block (mb) and 5 pixel
// blocks.  Its 18 weight A-fragments (3ky*3kx*2kb) are loaded ONCE into
// registers; the steady-state loop is ds_load_b128 + v_wmma only, with 5
// independent accumulators per tap for pipelining.  Rows are DMA'd by the
// Tensor Data Mover; compute on row ky overlaps the DMA of rows ky+1..
// OUTMODE: 0 = NHWC f16 (optional leaky-relu), 1 = NHWC f32, 2 = NCHW f32.
// ---------------------------------------------------------------------------
template <int COUT, bool LRELU, int OUTMODE>
__global__ __launch_bounds__((COUT / 16) * 2 * 32) void conv3x3_wmma(
    const _Float16* __restrict__ in,    // NHWC f16 [B][H][W][64]
    const _Float16* __restrict__ wp,    // packed A fragments
    void* __restrict__ outp) {
  constexpr int MB = COUT / 16;
  constexpr int NT = MB * 2 * 32;                   // threads per block
  constexpr int ROWH = (WW + 2) * CINC;             // halfs per padded row
  __shared__ __align__(16) _Float16 srow[3][ROWH];  // ~61 KB

  const int h = blockIdx.x;
  const int b = blockIdx.y;
  const int tid = threadIdx.x;
  const int lane = tid & 31;
  const int wv = tid >> 5;
  const int mb = wv % MB;                  // this wave's Cout block
  const int nbbase = (wv / MB) * 5;        // first of 5 pixel blocks
  const int hs = lane >> 4;                // half select
  const int khalf = hs ? 16 : 0;

  const bool inb0 = (h - 1) >= 0;
  const bool inb2 = (h + 1) < HH;
  // Wave 0 launches the row DMAs immediately (interior cols 1..160 only).
  if (wv == 0) {
    if (inb0)
      tdm_load_row(in + (size_t)(b * HH + h - 1) * WW * CINC,
                   (unsigned)(uintptr_t)(void*)&srow[0][CINC]);
    tdm_load_row(in + (size_t)(b * HH + h) * WW * CINC,
                 (unsigned)(uintptr_t)(void*)&srow[1][CINC]);
    if (inb2)
      tdm_load_row(in + (size_t)(b * HH + h + 1) * WW * CINC,
                   (unsigned)(uintptr_t)(void*)&srow[2][CINC]);
  }
  // outstanding DMAs allowed when consuming row r (in-order completion):
  int pend[3];
  pend[0] = 1 + (inb2 ? 1 : 0);
  pend[1] = inb2 ? 1 : 0;
  pend[2] = 0;

  // Load this wave's 18 weight fragments into registers (once).
  v16h warr[18];
#pragma unroll
  for (int t = 0; t < 18; ++t) {
    const int ky = t / 6, r = t % 6, kx = r >> 1, kb = r & 1;
    const int fr = ((ky * 3 + kx) * 2 + kb) * MB + mb;
    warr[t] = *(const v16h*)(wp + ((size_t)fr * 32 + lane) * 16);
  }

  // Zero-fill: column halo of every buffer, whole buffer for OOB rows.
  {
    const uint4 z = {0u, 0u, 0u, 0u};
    for (int r = 0; r < 3; ++r) {
      uint4* d4 = (uint4*)srow[r];
      const int gy = h + r - 1;
      if (gy < 0 || gy >= HH) {
        for (int i = tid; i < ROWH / 8; i += NT) d4[i] = z;
      } else if (tid < 16) {
        d4[(tid < 8) ? tid : ((WW + 1) * 8 + (tid - 8))] = z;
      }
    }
  }
  __syncthreads();                          // halo zeros visible to all waves

  v8f acc[5];
#pragma unroll
  for (int n = 0; n < 5; ++n)
    acc[n] = (v8f){0.f, 0.f, 0.f, 0.f, 0.f, 0.f, 0.f, 0.f};

  const int colb = (lane & 15) + khalf * 0;  // lane column within pixel block
#pragma unroll
  for (int ky = 0; ky < 3; ++ky) {
    if (wv == 0) wait_tensor_le(pend[ky]);   // row ky landed in LDS
    __syncthreads();
#pragma unroll
    for (int kx = 0; kx < 3; ++kx) {
#pragma unroll
      for (int kb = 0; kb < 2; ++kb) {
        const v16h aw = warr[ky * 6 + kx * 2 + kb];
        v16h bf[5];
#pragma unroll
        for (int n = 0; n < 5; ++n) {        // 5 independent B loads
          const int col = (nbbase + n) * 16 + (lane & 15) + kx;
          bf[n] = *(const v16h*)(&srow[ky][col * CINC + kb * 32 + khalf]);
        }
#pragma unroll
        for (int n = 0; n < 5; ++n)          // 5 independent WMMAs, shared A
          acc[n] = __builtin_amdgcn_wmma_f32_16x16x32_f16(
              false, aw, false, bf[n], (short)0, acc[n], false, false);
      }
    }
  }
  (void)colb;

  // D layout: lane = pixel (N), VGPR v = Cout row (M = v + 8*hs)
#pragma unroll
  for (int n = 0; n < 5; ++n) {
    const int pix = (nbbase + n) * 16 + (lane & 15);
    const size_t pbase = (size_t)(b * HH + h) * WW + pix;
    if constexpr (OUTMODE == 0) {
      _Float16* out = (_Float16*)outp;
      __align__(16) _Float16 tmp[8];
      for (int v = 0; v < 8; ++v) {
        float x = acc[n][v];
        if (LRELU) x = (x > 0.f) ? x : 0.1f * x;
        tmp[v] = (_Float16)x;
      }
      *(uint4*)(out + pbase * COUT + mb * 16 + hs * 8) = *(const uint4*)tmp;
    } else if constexpr (OUTMODE == 1) {
      float* out = (float*)outp;
      float* d = out + pbase * COUT + mb * 16 + hs * 8;
      for (int v = 0; v < 8; ++v) d[v] = acc[n][v];
    } else {
      float* out = (float*)outp;
      for (int v = 0; v < 8; ++v) {
        int o = mb * 16 + hs * 8 + v;
        out[((size_t)(b * COUT + o) * HH + h) * WW + pix] = acc[n][v];
      }
    }
  }
}

// ---------------------------------------------------------------------------
// Deformable bilinear sampling: hr NCHW f32, off NHWC f32 (ch 2c=dy, 2c+1=dx),
// out NHWC f16.
// ---------------------------------------------------------------------------
__global__ void bilinear_kernel(const float* __restrict__ hr,
                                const float* __restrict__ off,
                                _Float16* __restrict__ out) {
  int idx = blockIdx.x * blockDim.x + threadIdx.x;
  if (idx >= BB * HH * WW * CINC) return;
  int c = idx & 63;
  int p = idx >> 6;
  int w = p % WW;
  int t2 = p / WW;
  int h = t2 % HH;
  int b = t2 / HH;

  float oy = off[(size_t)p * 128 + 2 * c + 0];
  float ox = off[(size_t)p * 128 + 2 * c + 1];
  float cy = fminf(fmaxf((float)h + oy, 0.f), (float)(HH - 1));
  float cx = fminf(fmaxf((float)w + ox, 0.f), (float)(WW - 1));
  float y0f = floorf(cy), x0f = floorf(cx);
  int y0 = min(max((int)y0f, 0), HH - 1);
  int x0 = min(max((int)x0f, 0), WW - 1);
  int y1 = min(y0 + 1, HH - 1);
  int x1 = min(x0 + 1, WW - 1);
  float wy = cy - y0f, wx = cx - x0f;

  const float* pl = hr + ((size_t)b * CINC + c) * (HH * WW);
  float v00 = pl[y0 * WW + x0];
  float v01 = pl[y0 * WW + x1];
  float v10 = pl[y1 * WW + x0];
  float v11 = pl[y1 * WW + x1];
  float top = v00 * (1.f - wx) + v01 * wx;
  float bot = v10 * (1.f - wx) + v11 * wx;
  out[idx] = (_Float16)(top * (1.f - wy) + bot * wy);
}

// ---------------------------------------------------------------------------
extern "C" void kernel_launch(void* const* d_in, const int* in_sizes, int n_in,
                              void* d_out, int out_size, void* d_ws,
                              size_t ws_size, hipStream_t stream) {
  (void)in_sizes; (void)n_in; (void)out_size; (void)ws_size;
  // setup_inputs order: lr, hr, est_w1, est_w2, est_w3, offset_w, conv1_w
  const float* hr = (const float*)d_in[1];        // lr path is dead code
  const float* w1 = (const float*)d_in[2];
  const float* w2 = (const float*)d_in[3];
  const float* w3 = (const float*)d_in[4];
  const float* wo = (const float*)d_in[5];
  const float* wc = (const float*)d_in[6];
  float* out = (float*)d_out;

  char* ws = (char*)d_ws;
  const size_t WSZ64 = (size_t)9 * 2 * 4 * 32 * 16 * sizeof(_Float16);   // 72 KB
  const size_t WSZ128 = 2 * WSZ64;                                       // 144 KB
  _Float16* WP1 = (_Float16*)(ws + 0);
  _Float16* WP2 = (_Float16*)(ws + WSZ64);
  _Float16* WP3 = (_Float16*)(ws + 2 * WSZ64);
  _Float16* WPO = (_Float16*)(ws + 3 * WSZ64);
  _Float16* WPC = (_Float16*)(ws + 3 * WSZ64 + WSZ128);

  const size_t ACT = (size_t)BB * HH * WW * CINC * sizeof(_Float16);     // 25 MB
  const size_t A0 = 1u << 20;
  _Float16* bufA = (_Float16*)(ws + A0);              // hr NHWC f16
  _Float16* bufB = (_Float16*)(ws + A0 + ACT);
  _Float16* bufC = (_Float16*)(ws + A0 + 2 * ACT);
  float* offbuf = (float*)(ws + A0 + 3 * ACT);        // NHWC f32, 128 ch

  pack_w_kernel<<<(9 * 2 * 4 * 32 + 255) / 256, 256, 0, stream>>>(w1, WP1, 64);
  pack_w_kernel<<<(9 * 2 * 4 * 32 + 255) / 256, 256, 0, stream>>>(w2, WP2, 64);
  pack_w_kernel<<<(9 * 2 * 4 * 32 + 255) / 256, 256, 0, stream>>>(w3, WP3, 64);
  pack_w_kernel<<<(9 * 2 * 8 * 32 + 255) / 256, 256, 0, stream>>>(wo, WPO, 128);
  pack_w_kernel<<<(9 * 2 * 8 * 32 + 255) / 256, 256, 0, stream>>>(wc, WPC, 128);

  dim3 tg(WW / 32, HH, BB);
  nchw_to_nhwc_f16<<<tg, 256, 0, stream>>>(hr, bufA);

  dim3 cg(HH, BB);
  conv3x3_wmma<64, true, 0><<<cg, 256, 0, stream>>>(bufA, WP1, bufB);
  conv3x3_wmma<64, true, 0><<<cg, 256, 0, stream>>>(bufB, WP2, bufC);
  conv3x3_wmma<64, false, 0><<<cg, 256, 0, stream>>>(bufC, WP3, bufB);   // est_hr
  conv3x3_wmma<128, false, 1><<<cg, 512, 0, stream>>>(bufB, WPO, offbuf);

  const int NA = BB * HH * WW * CINC;
  bilinear_kernel<<<(NA + 255) / 256, 256, 0, stream>>>(hr, offbuf, bufC);

  conv3x3_wmma<128, false, 2><<<cg, 512, 0, stream>>>(bufC, WPC, out);
}
